// OptimalTransportLoss_84559316124427
// MI455X (gfx1250) — matrix operand order repaired and checked by reading, not compiled, not run on hardware
//
#include <hip/hip_runtime.h>
#include <hip/hip_bf16.h>
#include <math.h>

// ---------------------------------------------------------------------------
// Optimal-transport contrastive loss for MI455X (gfx1250).
//
// Roofline:
//  * cost GEMM: ~17 GFLOP fp16 -> v_wmma_f32_16x16x32_f16; cheap enough to
//    compute twice (minmax pass + sinkhorn pass) instead of spilling a 256 MB
//    cost tensor.
//  * sinkhorn (dominant): 100 iters x 4096 independent 128x128 matrices.
//    HBM-streaming would be >100 GB (>4.4 ms @ 23.3 TB/s); instead each
//    (i,j) matrix stays LDS-resident (320 KB/WGP) -> zero HBM bytes in loop.
//  * LDS peak = 64 banks x 4B = 256 B/clk, so a 32-lane b128 access (512 B)
//    costs >=2 cycles regardless of banking: full-width float4 passes with
//    stride 132 (16 B aligned rows) are bank-optimal AND 4x fewer LDS ops
//    than scalar conflict-free stride-133 walks. All hot passes use float4.
// ---------------------------------------------------------------------------

typedef _Float16 half_t;
typedef __attribute__((ext_vector_type(16))) _Float16 v16h;
typedef __attribute__((ext_vector_type(8)))  _Float16 v8h;
typedef __attribute__((ext_vector_type(8)))  float    v8f;

union V16H { v16h v; v8h h[2]; };

#define B64      64
#define N128     128
#define D128     128
#define CSTRIDE  132          // float stride: 132*4 = 528 B, 16 B aligned rows
#define HSTRIDE  136          // half stride for fp16 staging tiles
#define EPS_REG  0.1f
#define N_ITERS  100
#define STAB     1e-8f

// ---- block reduction helpers (256 threads) --------------------------------
__device__ __forceinline__ float block_reduce_sum(float v, float* red, int tid) {
  red[tid] = v; __syncthreads();
  for (int s = 128; s > 0; s >>= 1) {
    if (tid < s) red[tid] += red[tid + s];
    __syncthreads();
  }
  float r = red[0]; __syncthreads();
  return r;
}
__device__ __forceinline__ float block_reduce_min(float v, float* red, int tid) {
  red[tid] = v; __syncthreads();
  for (int s = 128; s > 0; s >>= 1) {
    if (tid < s) red[tid] = fminf(red[tid], red[tid + s]);
    __syncthreads();
  }
  float r = red[0]; __syncthreads();
  return r;
}
__device__ __forceinline__ float block_reduce_max(float v, float* red, int tid) {
  red[tid] = v; __syncthreads();
  for (int s = 128; s > 0; s >>= 1) {
    if (tid < s) red[tid] = fmaxf(red[tid], red[tid + s]);
    __syncthreads();
  }
  float r = red[0]; __syncthreads();
  return r;
}

// ---- stage a 128x128 f32 row-major global tile into LDS as fp16 -----------
__device__ __forceinline__ void stage_tile_fp16(const float* __restrict__ src,
                                                half_t* __restrict__ dst, int tid) {
  #pragma unroll 4
  for (int k = 0; k < 32; ++k) {
    int idx = tid + 256 * k;          // pair index 0..8191
    int m  = idx >> 6;                // row
    int dp = idx & 63;                // pair-within-row
    float2 f = ((const float2*)src)[m * 64 + dp];
    union { half_t h[2]; unsigned u; } p;
    p.h[0] = (half_t)f.x;
    p.h[1] = (half_t)f.y;
    *((unsigned*)(dst + m * HSTRIDE + 2 * dp)) = p.u;
  }
}

// ---- per-row squared norms of the fp16-rounded embeddings -----------------
__device__ __forceinline__ void sqnorms(const half_t* eh, const half_t* th,
                                        float* e2, float* t2, int tid) {
  const half_t* src = (tid < 128) ? eh : th;
  float*        dst = (tid < 128) ? e2 : t2;
  int m = tid & 127;
  const half_t* p = src + m * HSTRIDE;
  float s = 0.f;
  #pragma unroll 8
  for (int d = 0; d < D128; ++d) { float v = (float)p[d]; s += v * v; }
  dst[m] = s;
}

// ===========================================================================
// Kernel 0: init global min/max (cost >= 0 so uint ordering == float ordering)
// ===========================================================================
__global__ void ot_init_kernel(unsigned* gmm) {
  if (threadIdx.x == 0) { gmm[0] = 0x7F800000u /* +inf */; gmm[1] = 0u; }
}

// ===========================================================================
// Kernel 1: cost tile via WMMA, reduce global min/max
// ===========================================================================
__global__ void ot_cost_minmax_kernel(const float* __restrict__ eeg,
                                      const float* __restrict__ text,
                                      unsigned* __restrict__ gmm) {
  extern __shared__ char smem[];
  half_t* eh  = (half_t*)smem;                         // 34816 B
  half_t* th  = (half_t*)(smem + 34816);               // 34816 B
  float*  e2  = (float*)(smem + 69632);                // 512 B
  float*  t2  = e2 + 128;                              // 512 B
  float*  red = t2 + 128;                              // 1024 B

  const int tid = threadIdx.x;
  const int i = blockIdx.y, j = blockIdx.x;

  stage_tile_fp16(eeg  + (size_t)i * N128 * D128, eh, tid);
  stage_tile_fp16(text + (size_t)j * N128 * D128, th, tid);
  __syncthreads();
  sqnorms(eh, th, e2, t2, tid);
  __syncthreads();

  const int w = tid >> 5, lane = tid & 31;
  const int mb = w * 16;
  const int lrow = lane & 15, lhi = lane >> 4;

  // A fragments (16x32 f16): halves 0..7 = K kb+8*lhi, 8..15 = K kb+16+8*lhi.
  V16H afrag[4];
  #pragma unroll
  for (int kb4 = 0; kb4 < 4; ++kb4) {
    const half_t* base = eh + (mb + lrow) * HSTRIDE + kb4 * 32 + lhi * 8;
    afrag[kb4].h[0] = *(const v8h*)(base);
    afrag[kb4].h[1] = *(const v8h*)(base + 16);
  }

  float lmin = 3.4e38f, lmax = 0.f;
  for (int nb = 0; nb < N128; nb += 16) {
    // preload all 4 B fragments, then issue 4 chained WMMAs back-to-back
    V16H bfrag[4];
    #pragma unroll
    for (int kb4 = 0; kb4 < 4; ++kb4) {
      const half_t* bb = th + (nb + lrow) * HSTRIDE + kb4 * 32 + lhi * 8;
      bfrag[kb4].h[0] = *(const v8h*)(bb);
      bfrag[kb4].h[1] = *(const v8h*)(bb + 16);
    }
    v8f acc = {};
    #pragma unroll
    for (int kb4 = 0; kb4 < 4; ++kb4)
      acc = __builtin_amdgcn_wmma_f32_16x16x32_f16(
          false, afrag[kb4].v, false, bfrag[kb4].v, (short)0, acc, false, false);

    const float t2n = t2[nb + lrow];
    #pragma unroll
    for (int r = 0; r < 8; ++r) {
      int m = mb + r + lhi * 8;
      float sq = e2[m] + t2n - 2.0f * acc[r];
      float c  = __builtin_amdgcn_sqrtf(fmaxf(sq, 0.f));   // bare v_sqrt_f32
      lmin = fminf(lmin, c);
      lmax = fmaxf(lmax, c);
    }
  }

  float bmin = block_reduce_min(lmin, red, tid);
  float bmax = block_reduce_max(lmax, red, tid);
  if (tid == 0) {
    atomicMin(&gmm[0], __float_as_uint(bmin));
    atomicMax(&gmm[1], __float_as_uint(bmax));
  }
}

// ===========================================================================
// Kernel 2: normalized cost tile (WMMA) + 100 LDS-resident sinkhorn iters.
// LDS layout (144384 B):
//   [0)      C tile 128*132*4 = 67584
//   [67584)  overlay: {eh 34816 | th 34816} cost phase, T tile 67584 after
//   [137216) e2(512) t2(512) rsp(1024) csp(4096) red(1024)
// ===========================================================================
__global__ void ot_sinkhorn_kernel(const float* __restrict__ eeg,
                                   const float* __restrict__ text,
                                   const unsigned* __restrict__ gmm,
                                   float* __restrict__ distances) {
  extern __shared__ char smem[];
  float*  C   = (float*)smem;
  char*   Rg  = smem + 67584;
  half_t* eh  = (half_t*)Rg;
  half_t* th  = (half_t*)(Rg + 34816);
  float*  T   = (float*)Rg;                 // overlays eh/th once cost is done
  char*   S   = smem + 137216;
  float*  e2  = (float*)S;                  // 128 f
  float*  t2  = e2 + 128;                   // 128 f
  float*  rsp = t2 + 128;                   // [2][128] row-sum partials
  float*  csp = rsp + 256;                  // [8][128] col-sum partials (f4)
  float*  red = csp + 1024;                 // [256] reduce scratch

  const int tid = threadIdx.x;
  const int i = blockIdx.y, j = blockIdx.x;

  // ---- phase 1: normalized cost tile into C -------------------------------
  stage_tile_fp16(eeg  + (size_t)i * N128 * D128, eh, tid);
  stage_tile_fp16(text + (size_t)j * N128 * D128, th, tid);
  __syncthreads();
  sqnorms(eh, th, e2, t2, tid);
  __syncthreads();

  const float gmin = __uint_as_float(gmm[0]);
  const float gmax = __uint_as_float(gmm[1]);
  const float ginv = __builtin_amdgcn_rcpf(gmax - gmin);

  const int w = tid >> 5, lane = tid & 31;
  const int mb = w * 16;
  const int lrow = lane & 15, lhi = lane >> 4;

  V16H afrag[4];
  #pragma unroll
  for (int kb4 = 0; kb4 < 4; ++kb4) {
    const half_t* base = eh + (mb + lrow) * HSTRIDE + kb4 * 32 + lhi * 8;
    afrag[kb4].h[0] = *(const v8h*)(base);
    afrag[kb4].h[1] = *(const v8h*)(base + 16);
  }
  for (int nb = 0; nb < N128; nb += 16) {
    V16H bfrag[4];
    #pragma unroll
    for (int kb4 = 0; kb4 < 4; ++kb4) {
      const half_t* bb = th + (nb + lrow) * HSTRIDE + kb4 * 32 + lhi * 8;
      bfrag[kb4].h[0] = *(const v8h*)(bb);
      bfrag[kb4].h[1] = *(const v8h*)(bb + 16);
    }
    v8f acc = {};
    #pragma unroll
    for (int kb4 = 0; kb4 < 4; ++kb4)
      acc = __builtin_amdgcn_wmma_f32_16x16x32_f16(
          false, afrag[kb4].v, false, bfrag[kb4].v, (short)0, acc, false, false);

    const int n = nb + lrow;
    const float t2n = t2[n];
    #pragma unroll
    for (int r = 0; r < 8; ++r) {
      int m = mb + r + lhi * 8;
      float sq = e2[m] + t2n - 2.0f * acc[r];
      float c  = __builtin_amdgcn_sqrtf(fmaxf(sq, 0.f));
      C[m * CSTRIDE + n] = (c - gmin) * ginv;    // two disjoint 16-bank groups
    }
  }
  __syncthreads();   // eh/th dead from here; T overlays that region

  // Vectorized full-tile mapping: quad index l4 = tid + 256*k (k<16):
  //   r = l4>>5 (same for all 32 lanes of a wave), cq = l4&31 -> col 4*cq.
  // Each float4 access is 16 B aligned (132*4 and 16*cq both 16 B multiples).

  // ---- phase 2: T = exp(-C/reg) / sum(T) ----------------------------------
  float lsum = 0.f;
  #pragma unroll 4
  for (int k = 0; k < 16; ++k) {
    int l4 = tid + 256 * k, r = l4 >> 5, cq = l4 & 31;
    float4 cv = ((const float4*)(C + r * CSTRIDE))[cq];
    float4 tv;
    tv.x = __expf(cv.x * (-1.0f / EPS_REG));
    tv.y = __expf(cv.y * (-1.0f / EPS_REG));
    tv.z = __expf(cv.z * (-1.0f / EPS_REG));
    tv.w = __expf(cv.w * (-1.0f / EPS_REG));
    ((float4*)(T + r * CSTRIDE))[cq] = tv;
    lsum += (tv.x + tv.y) + (tv.z + tv.w);
  }
  float tot = block_reduce_sum(lsum, red, tid);
  const float tsc = __builtin_amdgcn_rcpf(tot);
  #pragma unroll 4
  for (int k = 0; k < 16; ++k) {
    int l4 = tid + 256 * k, r = l4 >> 5, cq = l4 & 31;
    float4 tv = ((const float4*)(T + r * CSTRIDE))[cq];
    tv.x *= tsc; tv.y *= tsc; tv.z *= tsc; tv.w *= tsc;
    ((float4*)(T + r * CSTRIDE))[cq] = tv;
  }
  __syncthreads();

  // ---- phase 3: 100 sinkhorn iterations, all float4 LDS traffic -----------
  const int rr = tid & 127, hh = tid >> 7;   // row-sum split: 2 threads/row
  const int cq = tid & 31,  sl = tid >> 5;   // col-sum split: quad x 16-row slice
  #pragma unroll 1
  for (int it = 0; it < N_ITERS; ++it) {
    // row sums (axis 3): 16 float4 loads per thread
    {
      const float4* p = (const float4*)(T + rr * CSTRIDE + hh * 64);
      float4 a = p[0];
      #pragma unroll
      for (int k = 1; k < 16; ++k) {
        float4 v = p[k];
        a.x += v.x; a.y += v.y; a.z += v.z; a.w += v.w;
      }
      rsp[hh * 128 + rr] = (a.x + a.y) + (a.z + a.w);
    }
    __syncthreads();
    if (tid < 128)
      rsp[tid] = __builtin_amdgcn_rcpf(rsp[tid] + rsp[128 + tid] + STAB);
    __syncthreads();
    #pragma unroll 4
    for (int k = 0; k < 16; ++k) {            // whole wave shares r: broadcast
      int l4 = tid + 256 * k, r = l4 >> 5, q = l4 & 31;
      float f = rsp[r];
      float4 tv = ((const float4*)(T + r * CSTRIDE))[q];
      tv.x *= f; tv.y *= f; tv.z *= f; tv.w *= f;
      ((float4*)(T + r * CSTRIDE))[q] = tv;
    }
    __syncthreads();
    // col sums (axis 2): thread sums 16 rows of its column-quad
    {
      const float* base = T + (sl * 16) * CSTRIDE;
      float4 a = ((const float4*)base)[cq];
      #pragma unroll
      for (int k = 1; k < 16; ++k) {
        float4 v = ((const float4*)(base + k * CSTRIDE))[cq];
        a.x += v.x; a.y += v.y; a.z += v.z; a.w += v.w;
      }
      ((float4*)csp)[sl * 32 + cq] = a;       // csp[sl][4cq..4cq+3]
    }
    __syncthreads();
    if (tid < 128) {                          // combine 8 slices -> 1/colsum
      float s = csp[tid] + csp[128 + tid];
      #pragma unroll
      for (int k = 2; k < 8; ++k) s += csp[k * 128 + tid];
      csp[tid] = __builtin_amdgcn_rcpf(s + STAB);
    }
    __syncthreads();
    #pragma unroll 4
    for (int k = 0; k < 16; ++k) {
      int l4 = tid + 256 * k, r = l4 >> 5, q = l4 & 31;
      float4 f = ((const float4*)csp)[q];
      float4 tv = ((const float4*)(T + r * CSTRIDE))[q];
      tv.x *= f.x; tv.y *= f.y; tv.z *= f.z; tv.w *= f.w;
      ((float4*)(T + r * CSTRIDE))[q] = tv;
    }
    __syncthreads();
  }

  // ---- phase 4: distance = sum(T * C) -------------------------------------
  float ld = 0.f;
  #pragma unroll 4
  for (int k = 0; k < 16; ++k) {
    int l4 = tid + 256 * k, r = l4 >> 5, q = l4 & 31;
    float4 tv = ((const float4*)(T + r * CSTRIDE))[q];
    float4 cv = ((const float4*)(C + r * CSTRIDE))[q];
    ld += tv.x * cv.x + tv.y * cv.y + tv.z * cv.z + tv.w * cv.w;
  }
  float d = block_reduce_sum(ld, red, tid);
  if (tid == 0) distances[i * B64 + j] = d;
}

// ===========================================================================
// Kernel 3: similarity + cross-entropy loss (single block)
//   out[0] = loss, out[1..4096] = similarity row-major
// ===========================================================================
__global__ void ot_finalize_kernel(const float* __restrict__ distances,
                                   float* __restrict__ out) {
  __shared__ float red[256];
  __shared__ float sim[64 * 65];
  const int tid = threadIdx.x;

  float lmin = 3.4e38f, lmax = -3.4e38f;
  #pragma unroll
  for (int k = 0; k < 16; ++k) {
    float v = distances[tid + 256 * k];
    lmin = fminf(lmin, v); lmax = fmaxf(lmax, v);
  }
  float dmn = block_reduce_min(lmin, red, tid);
  float dmx = block_reduce_max(lmax, red, tid);
  const float inv = 1.0f / (dmx - dmn);

  #pragma unroll
  for (int k = 0; k < 16; ++k) {
    int idx = tid + 256 * k, r = idx >> 6, c = idx & 63;
    float s = __expf(-(distances[idx] - dmn) * inv);
    sim[r * 65 + c] = s;
    out[1 + idx] = s;
  }
  __syncthreads();

  float lp = 0.f;
  if (tid < 64) {
    float mx = -3.4e38f;
    #pragma unroll 8
    for (int c = 0; c < 64; ++c) mx = fmaxf(mx, sim[tid * 65 + c]);
    float se = 0.f;
    #pragma unroll 8
    for (int c = 0; c < 64; ++c) se += __expf(sim[tid * 65 + c] - mx);
    lp = sim[tid * 65 + tid] - mx - logf(se);
  }
  float s = block_reduce_sum((tid < 64) ? lp : 0.f, red, tid);
  if (tid == 0) out[0] = -s / 64.0f;
}

// ===========================================================================
extern "C" void kernel_launch(void* const* d_in, const int* in_sizes, int n_in,
                              void* d_out, int out_size, void* d_ws, size_t ws_size,
                              hipStream_t stream) {
  (void)in_sizes; (void)n_in; (void)out_size; (void)ws_size;
  const float* eeg  = (const float*)d_in[0];
  const float* text = (const float*)d_in[1];
  float* out = (float*)d_out;

  unsigned* gmm  = (unsigned*)d_ws;                    // 2 x u32 min/max bits
  float*    dist = (float*)((char*)d_ws + 16);         // 4096 f32 distances

  const size_t shmem1 = 34816u * 2 + (128 + 128 + 256) * 4;                  // 71680
  const size_t shmem2 = 67584u + 69632u + (128 + 128 + 256 + 1024 + 256) * 4; // 144384

  hipFuncSetAttribute((const void*)ot_cost_minmax_kernel,
                      hipFuncAttributeMaxDynamicSharedMemorySize, (int)shmem1);
  hipFuncSetAttribute((const void*)ot_sinkhorn_kernel,
                      hipFuncAttributeMaxDynamicSharedMemorySize, (int)shmem2);

  dim3 grid(B64, B64);
  ot_init_kernel<<<1, 32, 0, stream>>>(gmm);
  ot_cost_minmax_kernel<<<grid, 256, shmem1, stream>>>(eeg, text, gmm);
  ot_sinkhorn_kernel<<<grid, 256, shmem2, stream>>>(eeg, text, gmm, dist);
  ot_finalize_kernel<<<1, 256, 0, stream>>>(dist, out);
}
